// BeamSearchSequenceGenerator_38336878084624
// MI455X (gfx1250) — matrix operand order repaired and checked by reading, not compiled, not run on hardware
//
#include <hip/hip_runtime.h>
#include <hip/hip_bf16.h>
#include <math.h>

// ---------------------------------------------------------------------------
// Beam search decoder for MI455X (gfx1250, wave32, WMMA).
//
// Roofline: per step logits GEMM = 32x1024x32000 (2.1 GFLOP) against W_out
// (131 MB fp32).  W_out fits in the 192 MB L2, so steps 2..31 stream weights
// from L2 -> compute-bound -> use v_wmma_f32_16x16x32_bf16 (bf16 in, f32 acc).
// B is converted f32->bf16 on the fly while staging transposed into LDS
// (packed k-pair ds_store_b32), so no 65 MB weight copy is needed in ws.
// ---------------------------------------------------------------------------

#define BB   8          // batch
#define KB   4          // beams
#define NR   (BB*KB)    // 32 rows
#define DD   1024
#define VV   32000
#define TT   32         // max seq len
#define PADT 0
#define BOST 1
#define EOST 2

typedef __attribute__((ext_vector_type(2)))  __bf16 v2bf;
typedef __attribute__((ext_vector_type(8)))  __bf16 v8bf;
typedef __attribute__((ext_vector_type(16))) __bf16 v16bf;
typedef __attribute__((ext_vector_type(8)))  float  v8f;

__device__ __forceinline__ v16bf cat8(v8bf lo, v8bf hi) {
  v16bf r;
#pragma unroll
  for (int i = 0; i < 8; ++i) { r[i] = lo[i]; r[i + 8] = hi[i]; }
  return r;
}

// ---------------------------------------------------------------------------
// Token init
// ---------------------------------------------------------------------------
__global__ void set_tokens_bos(int* tokens) {
  if (threadIdx.x < NR) tokens[threadIdx.x] = BOST;
}

// ---------------------------------------------------------------------------
// Gather embedding rows for the 32 live tokens, convert fp32 -> bf16.
// ---------------------------------------------------------------------------
__global__ __launch_bounds__(256)
void gather_embed(const float* __restrict__ emb, const int* __restrict__ tokens,
                  __bf16* __restrict__ out) {
  const int r = blockIdx.x;
  const int tok = tokens[r];
  const float* src = emb + (size_t)tok * DD;
  __bf16* dst = out + (size_t)r * DD;
  for (int i = threadIdx.x; i < DD; i += 256) dst[i] = (__bf16)src[i];
}

// ---------------------------------------------------------------------------
// WMMA GEMM: C[32 x N] = A_bf16[32 x Kdim] @ B_f32[Kdim x N]
//   MODE 0: outF[row*N+col] = acc                       (logits)
//   MODE 1: outH[row*N+col] = bf16(tanh(acc+bias[col])) (decoder hidden)
//
// 256 threads = 8 waves.  Workgroup tile = 32 x 128.  Wave w: mtile = w&1,
// n-tile pair = w>>1 -> two 16x16 tiles / two accumulators per wave (A
// fragment reused).  Per 32-deep K chunk the B block (32k x 128n) is staged
// to LDS transposed: each staging thread loads two consecutive K rows x 8
// cols, converts (v_cvt_pk_bf16_f32) and stores packed (k,k+1) bf16 pairs
// with ds_store_b32.  Padded stride 40 elem (80 B) keeps 16 B fragment
// alignment and skews banks.
// ---------------------------------------------------------------------------
template <int MODE>
__global__ __launch_bounds__(256)
void gemm_tn(const __bf16* __restrict__ A, const float* __restrict__ B,
             int Kdim, int N, float* __restrict__ outF,
             __bf16* __restrict__ outH, const float* __restrict__ bias) {
  constexpr int LDSB = 40;                    // padded K stride (elements)
  __shared__ __bf16 ldsB[128 * LDSB];         // 10240 bytes

  const int tid   = threadIdx.x;
  const int lane  = tid & 31;
  const int wave  = tid >> 5;                 // 0..7
  const int mtile = wave & 1;                 // 0..1
  const int npair = wave >> 1;                // 0..3  -> n-tiles 2*npair, +1
  const int half  = lane >> 4;                // 0..1
  const int l15   = lane & 15;
  const int nbase = blockIdx.x * 128;

  // staging indices: thread covers K rows (2*kp, 2*kp+1) x 8 columns
  const int s_k  = (tid >> 4) * 2;            // 0,2,..,30
  const int s_n8 = (tid & 15) * 8;            // 0..120

  v8f acc0 = {};
  v8f acc1 = {};

  const __bf16* arow = A + (size_t)(mtile * 16 + l15) * Kdim;
  const int ncol0 = (2 * npair) * 16 + l15;   // local col, tile 0
  const int ncol1 = ncol0 + 16;               // local col, tile 1

  for (int k0 = 0; k0 < Kdim; k0 += 32) {
    // ---- stage B[k0+s_k .. +1][nbase+s_n8 .. +7] -> ldsB[n][k] (bf16 pairs)
    const float* src0 = B + (size_t)(k0 + s_k) * N + nbase + s_n8;
    const float* src1 = src0 + N;
    float4 a0 = *(const float4*)(src0);
    float4 a1 = *(const float4*)(src0 + 4);
    float4 b0 = *(const float4*)(src1);
    float4 b1 = *(const float4*)(src1 + 4);
    float r0[8] = {a0.x, a0.y, a0.z, a0.w, a1.x, a1.y, a1.z, a1.w};
    float r1[8] = {b0.x, b0.y, b0.z, b0.w, b1.x, b1.y, b1.z, b1.w};
#pragma unroll
    for (int i = 0; i < 8; ++i) {
      v2bf p = {(__bf16)r0[i], (__bf16)r1[i]};          // packs k, k+1
      *(v2bf*)(&ldsB[(s_n8 + i) * LDSB + s_k]) = p;     // one ds_store_b32
    }
    __syncthreads();

    // ---- A fragment (ISA 16-bit A 16x32 layout):
    // lanes 0-15: elems 0..7 = K k0+0..7,  elems 8..15 = K k0+16..23
    // lanes16-31: elems 0..7 = K k0+8..15, elems 8..15 = K k0+24..31
    v8bf alo = *(const v8bf*)(arow + k0 + 8 * half);
    v8bf ahi = *(const v8bf*)(arow + k0 + 16 + 8 * half);
    v16bf av = cat8(alo, ahi);

    // ---- B fragments: lane covers column n, elems 0..7 = K k0+8*half+e, elems 8..15 = K k0+16+8*half+(e-8)
    const __bf16* bc0 = &ldsB[ncol0 * LDSB + 8 * half];
    const __bf16* bc1 = &ldsB[ncol1 * LDSB + 8 * half];
    v16bf bv0 = cat8(*(const v8bf*)(bc0), *(const v8bf*)(bc0 + 16));
    v16bf bv1 = cat8(*(const v8bf*)(bc1), *(const v8bf*)(bc1 + 16));

    acc0 = __builtin_amdgcn_wmma_f32_16x16x32_bf16(
        false, av, false, bv0, (short)0, acc0, false, false);
    acc1 = __builtin_amdgcn_wmma_f32_16x16x32_bf16(
        false, av, false, bv1, (short)0, acc1, false, false);
    __syncthreads();
  }

  // ---- epilogue.  C layout: VGPR j, lanes 0-15 -> M=j; lanes 16-31 -> M=j+8
  const int col0 = nbase + ncol0;
  const int col1 = nbase + ncol1;
#pragma unroll
  for (int j = 0; j < 8; ++j) {
    const int row = mtile * 16 + j + 8 * half;
    if (MODE == 0) {
      outF[(size_t)row * N + col0] = acc0[j];
      outF[(size_t)row * N + col1] = acc1[j];
    } else {
      outH[(size_t)row * N + col0] = (__bf16)tanhf(acc0[j] + bias[col0]);
      outH[(size_t)row * N + col1] = (__bf16)tanhf(acc1[j] + bias[col1]);
    }
  }
}

// ---------------------------------------------------------------------------
// Per-row online logsumexp + top-4 (value,index) over logits[row][0..N).
// One 256-thread workgroup per row.
// ---------------------------------------------------------------------------
__device__ __forceinline__ bool better(float v1, int i1, float v2, int i2) {
  return (v1 > v2) || (v1 == v2 && i1 < i2);
}

__global__ __launch_bounds__(256)
void row_reduce(const float* __restrict__ logits, int N,
                float* __restrict__ lse, float* __restrict__ t4v,
                int* __restrict__ t4i) {
  __shared__ float sm[256];
  __shared__ float ss[256];
  __shared__ float sv[256 * 4];
  __shared__ int   si[256 * 4];

  const int row = blockIdx.x;
  const int tid = threadIdx.x;
  const float* r = logits + (size_t)row * N;

  float m = -INFINITY, s = 0.f;
  float tv[4] = {-INFINITY, -INFINITY, -INFINITY, -INFINITY};
  int   ti[4] = {0x7fffffff, 0x7fffffff, 0x7fffffff, 0x7fffffff};

  for (int i = tid; i < N; i += 256) {
    const float v = r[i];
    if (v > m) { s = s * expf(m - v) + 1.f; m = v; }
    else       { s += expf(v - m); }
    if (better(v, i, tv[3], ti[3])) {
      tv[3] = v; ti[3] = i;
#pragma unroll
      for (int j = 3; j > 0; --j) {
        if (better(tv[j], ti[j], tv[j - 1], ti[j - 1])) {
          float fv = tv[j]; tv[j] = tv[j - 1]; tv[j - 1] = fv;
          int fi = ti[j];   ti[j] = ti[j - 1]; ti[j - 1] = fi;
        }
      }
    }
  }
  sm[tid] = m; ss[tid] = s;
#pragma unroll
  for (int j = 0; j < 4; ++j) { sv[tid * 4 + j] = tv[j]; si[tid * 4 + j] = ti[j]; }
  __syncthreads();

  for (int stride = 128; stride > 0; stride >>= 1) {
    if (tid < stride) {
      // logsumexp combine
      float ma = sm[tid], mb = sm[tid + stride];
      float sa = ss[tid], sb = ss[tid + stride];
      if (mb > ma) { float t = ma; ma = mb; mb = t; t = sa; sa = sb; sb = t; }
      ss[tid] = sa + sb * expf(mb - ma);
      sm[tid] = ma;
      // top4 merge (two sorted lists, two-pointer, tie -> lower index)
      float av4[4], bv4[4], rv[4];
      int   ai4[4], bi4[4], ri[4];
#pragma unroll
      for (int j = 0; j < 4; ++j) {
        av4[j] = sv[tid * 4 + j];            ai4[j] = si[tid * 4 + j];
        bv4[j] = sv[(tid + stride) * 4 + j]; bi4[j] = si[(tid + stride) * 4 + j];
      }
      int x = 0, y = 0;
#pragma unroll
      for (int o = 0; o < 4; ++o) {
        if (better(av4[x], ai4[x], bv4[y], bi4[y])) { rv[o] = av4[x]; ri[o] = ai4[x]; ++x; }
        else                                        { rv[o] = bv4[y]; ri[o] = bi4[y]; ++y; }
      }
#pragma unroll
      for (int j = 0; j < 4; ++j) { sv[tid * 4 + j] = rv[j]; si[tid * 4 + j] = ri[j]; }
    }
    __syncthreads();
  }

  if (tid == 0) {
    lse[row] = sm[0] + logf(ss[0]);
#pragma unroll
    for (int j = 0; j < 4; ++j) { t4v[row * 4 + j] = sv[j]; t4i[row * 4 + j] = si[j]; }
  }
}

// ---------------------------------------------------------------------------
// Step-0 beam init from the BOS logits (top-4 per batch row = initial beams).
// ---------------------------------------------------------------------------
__global__ void init_beams(const float* __restrict__ t4v, const int* __restrict__ t4i,
                           const float* __restrict__ lse, float* scores, int* finished,
                           float* lengths, int* seqs, int* tokens) {
  const int t = threadIdx.x;
  if (t < NR) {
    const int b = t >> 2, k = t & 3;
    const int row = b * KB;                 // all rows of a batch are identical at step 0
    const int tok = t4i[row * 4 + k];
    scores[t]   = t4v[row * 4 + k] - lse[row];
    finished[t] = (tok == EOST) ? 1 : 0;
    lengths[t]  = 1.0f;
    tokens[t]   = tok;
    int* s = seqs + t * TT;
    s[0] = BOST; s[1] = tok;
    for (int i = 2; i < TT; ++i) s[i] = PADT;
  }
}

// ---------------------------------------------------------------------------
// One beam-search step at sequence position tpos:
// merge per-row top-4 (unfinished) / pad candidate (finished), top-K over the
// union, then gather-reorder all per-beam state.  Single workgroup.
// ---------------------------------------------------------------------------
__global__ __launch_bounds__(256)
void beam_update(int tpos, const float* __restrict__ t4v, const int* __restrict__ t4i,
                 const float* __restrict__ lse, float* scores, int* finished,
                 float* lengths, int* seqs, int* tokens) {
  __shared__ int   oldseq[NR * TT];
  __shared__ float oldlen[NR];
  __shared__ int   oldfin[NR];
  __shared__ float oldsc[NR];
  __shared__ int   nbeam[NR];
  __shared__ int   ntok[NR];
  __shared__ float nsc[NR];

  const int tid = threadIdx.x;
  for (int i = tid; i < NR * TT; i += 256) oldseq[i] = seqs[i];
  if (tid < NR) { oldlen[tid] = lengths[tid]; oldfin[tid] = finished[tid]; oldsc[tid] = scores[tid]; }
  __syncthreads();

  if (tid < BB) {
    const int b = tid;
    float cv[16]; int ci[16], cb[16], ct[16];
    int nc = 0;
    for (int k = 0; k < KB; ++k) {
      const int r = b * KB + k;
      if (oldfin[r]) {                       // only candidate: <pad> at +0
        cv[nc] = oldsc[r]; ci[nc] = k * VV + PADT; cb[nc] = k; ct[nc] = PADT; ++nc;
      } else {
        const float L = lse[r];
        for (int j = 0; j < 4; ++j) {
          const int tk = t4i[r * 4 + j];
          cv[nc] = oldsc[r] + t4v[r * 4 + j] - L;
          ci[nc] = k * VV + tk; cb[nc] = k; ct[nc] = tk; ++nc;
        }
      }
    }
    for (int sel = 0; sel < KB; ++sel) {     // selection of top-K (tie: lower idx)
      int best = sel;
      for (int j = sel + 1; j < nc; ++j)
        if (better(cv[j], ci[j], cv[best], ci[best])) best = j;
      float fv = cv[sel]; cv[sel] = cv[best]; cv[best] = fv;
      int f1 = ci[sel]; ci[sel] = ci[best]; ci[best] = f1;
      f1 = cb[sel]; cb[sel] = cb[best]; cb[best] = f1;
      f1 = ct[sel]; ct[sel] = ct[best]; ct[best] = f1;
      nsc[b * KB + sel] = cv[sel]; nbeam[b * KB + sel] = cb[sel]; ntok[b * KB + sel] = ct[sel];
    }
  }
  __syncthreads();

  if (tid < NR) {
    const int b = tid >> 2;
    const int src = b * KB + nbeam[tid];
    float ln = oldlen[src];
    int fin = oldfin[src];
    ln += fin ? 0.f : 1.f;                   // lengths += ~finished (pre-update)
    const int tok = ntok[tid];
    fin = fin | ((tok == EOST) ? 1 : 0);
    lengths[tid] = ln; finished[tid] = fin; scores[tid] = nsc[tid]; tokens[tid] = tok;
  }
  for (int i = tid; i < NR * TT; i += 256) { // seqs gather + set position tpos
    const int bk = i >> 5, pos = i & 31;
    const int b = bk >> 2;
    const int src = b * KB + nbeam[bk];
    seqs[i] = (pos == tpos) ? ntok[bk] : oldseq[src * TT + pos];
  }
}

// ---------------------------------------------------------------------------
// GNMT length penalty + output packing: [seqs (1024 floats), scores (32)].
// ---------------------------------------------------------------------------
__global__ void finalize(const int* __restrict__ seqs, const float* __restrict__ scores,
                         const float* __restrict__ lengths, float* __restrict__ out) {
  const int i = blockIdx.x * 256 + threadIdx.x;
  if (i < NR * TT) out[i] = (float)seqs[i];
  if (i < NR) {
    const float lp = powf((5.0f + lengths[i]) / 6.0f, 0.6f);
    out[NR * TT + i] = scores[i] / lp;
  }
}

// ---------------------------------------------------------------------------
// Host orchestration (stream-ordered, graph-capture safe).
// ---------------------------------------------------------------------------
extern "C" void kernel_launch(void* const* d_in, const int* in_sizes, int n_in,
                              void* d_out, int out_size, void* d_ws, size_t ws_size,
                              hipStream_t stream) {
  (void)in_sizes; (void)n_in; (void)out_size; (void)ws_size;
  const float* emb = (const float*)d_in[0];   // [V, D]
  const float* Wd  = (const float*)d_in[1];   // [D, D]
  const float* bd  = (const float*)d_in[2];   // [D]
  const float* Wo  = (const float*)d_in[3];   // [D, V]

  size_t off = 0;
  char* base = (char*)d_ws;
  auto carve = [&](size_t bytes) -> void* {
    void* p = base + off;
    off += (bytes + 255) & ~(size_t)255;
    return p;
  };
  float*  logits = (float*) carve((size_t)NR * VV * sizeof(float));   // 4.1 MB
  __bf16* embr   = (__bf16*)carve((size_t)NR * DD * sizeof(__bf16));
  __bf16* hbf    = (__bf16*)carve((size_t)NR * DD * sizeof(__bf16));
  float*  lse    = (float*) carve(NR * sizeof(float));
  float*  t4v    = (float*) carve(NR * 4 * sizeof(float));
  int*    t4i    = (int*)   carve(NR * 4 * sizeof(int));
  int*    tokens = (int*)   carve(NR * sizeof(int));
  float*  scores = (float*) carve(NR * sizeof(float));
  int*    fin    = (int*)   carve(NR * sizeof(int));
  float*  lens   = (float*) carve(NR * sizeof(float));
  int*    seqs   = (int*)   carve(NR * TT * sizeof(int));

  set_tokens_bos<<<1, 32, 0, stream>>>(tokens);

  for (int step = 0; step < TT - 1; ++step) {       // 31 decode steps
    gather_embed<<<NR, 256, 0, stream>>>(emb, tokens, embr);
    // h = tanh(emb @ W_dec + b)   (32x1024x1024, bf16 WMMA, tanh epilogue)
    gemm_tn<1><<<DD / 128, 256, 0, stream>>>(embr, Wd, DD, DD, nullptr, hbf, bd);
    // logits = h @ W_out          (32x1024x32000, bf16 WMMA, fp32 out)
    gemm_tn<0><<<VV / 128, 256, 0, stream>>>(hbf, Wo, DD, VV, logits, nullptr, nullptr);
    row_reduce<<<NR, 256, 0, stream>>>(logits, VV, lse, t4v, t4i);
    if (step == 0)
      init_beams<<<1, 32, 0, stream>>>(t4v, t4i, lse, scores, fin, lens, seqs, tokens);
    else
      beam_update<<<1, 256, 0, stream>>>(step + 1, t4v, t4i, lse, scores, fin, lens,
                                         seqs, tokens);
  }
  finalize<<<(NR * TT + NR + 255) / 256 + 1, 256, 0, stream>>>(seqs, scores, lens,
                                                               (float*)d_out);
}
